// Decoder_29214367547665
// MI455X (gfx1250) — matrix operand order repaired and checked by reading, not compiled, not run on hardware
//
#include <hip/hip_runtime.h>
#include <hip/hip_bf16.h>
#include <stdint.h>

// Problem constants (match reference)
#define B_  64
#define P_  196
#define D_  2048
#define E_  512
#define V_  10000
#define T1_ 33
#define T_  32   // T1-1

typedef __attribute__((ext_vector_type(16))) __bf16 v16bf;
typedef __attribute__((ext_vector_type(8)))  float  v8f;
typedef __attribute__((ext_vector_type(4)))  int    v4i_;

__device__ __forceinline__ unsigned short f32_bf16(float f) {
  unsigned u = __float_as_uint(f);
  u += 0x7fffu + ((u >> 16) & 1u);          // round-to-nearest-even
  return (unsigned short)(u >> 16);
}
__device__ __forceinline__ float sigmoidf_(float x) { return 1.f / (1.f + __expf(-x)); }

union FragB { uint4 u[2]; v16bf v; };

// ---------------------------------------------------------------------------
// Async global->LDS copy (CDNA5): 16B per lane, tracked by ASYNCcnt.
// Builtin parameter types (from hipcc diagnostic): v4i pointers, global + LDS.
// ---------------------------------------------------------------------------
#if __has_builtin(__builtin_amdgcn_global_load_async_to_lds_b128)
#define HAVE_ASYNC_LDS 1
#else
#define HAVE_ASYNC_LDS 0
#endif

__device__ __forceinline__ void copy16_to_lds(unsigned short* lds_dst,
                                              const unsigned short* gsrc) {
#if HAVE_ASYNC_LDS
  __builtin_amdgcn_global_load_async_to_lds_b128(
      (__attribute__((address_space(1))) v4i_*)(v4i_*)gsrc,
      (__attribute__((address_space(3))) v4i_*)(v4i_*)lds_dst,
      0, 0);
#else
  *(uint4*)lds_dst = *(const uint4*)gsrc;
#endif
}

__device__ __forceinline__ void wait_async_copies() {
#if HAVE_ASYNC_LDS
#if __has_builtin(__builtin_amdgcn_s_wait_asynccnt)
  __builtin_amdgcn_s_wait_asynccnt(0);
#else
  asm volatile("s_wait_asynccnt 0x0" ::: "memory");
#endif
#endif
}

// ---------------------------------------------------------------------------
// One-time weight prep: W (N x K, f32, row-major) -> Wt (K x N, bf16, K-major)
// Classic LDS tiled transpose, coalesced on both sides.
// ---------------------------------------------------------------------------
__global__ __launch_bounds__(256) void transpose_bf16_kernel(
    const float* __restrict__ W, unsigned short* __restrict__ Wt, int N, int K)
{
  __shared__ float tile[32][33];
  const int k0 = blockIdx.x * 32, n0 = blockIdx.y * 32;
  const int tx = threadIdx.x & 31, ty = threadIdx.x >> 5;   // 32 x 8
  #pragma unroll
  for (int i = ty; i < 32; i += 8) {
    int n = n0 + i, k = k0 + tx;
    tile[i][tx] = (n < N && k < K) ? W[(size_t)n * K + k] : 0.f;
  }
  __syncthreads();
  #pragma unroll
  for (int i = ty; i < 32; i += 8) {
    int k = k0 + i, n = n0 + tx;
    if (k < K && n < N) Wt[(size_t)k * N + n] = f32_bf16(tile[tx][i]);
  }
}

// ---------------------------------------------------------------------------
// WMMA GEMM:  C[M,N](ldc) = act( A[M,K](lda,f32) @ W^T + bias ) (+= if accum)
// W is supplied pre-transposed & pre-converted: Wt = bf16 (K x N, K-major).
// Tile: 64x64x64, 256 threads = 8 waves; wave computes 16(M)x32(N);
// 4 x v_wmma_f32_16x16x32_bf16 per sync period. B tile staged via
// global_load_async_to_lds_b128 (ASYNCcnt), A tile converted f32->bf16 inline.
// ---------------------------------------------------------------------------
#define BM 64
#define BN 64
#define BK 64
#define AP 72   // As row pitch (bf16 elems): 144B, 16B-aligned rows
#define WP 72   // Ws row pitch (bf16 elems): 144B, 16B-aligned rows

__global__ __launch_bounds__(256) void wmma_gemm_xwt(
    const float* __restrict__ A, int lda,
    const unsigned short* __restrict__ Wt,   // K x N bf16, K-major
    const float* __restrict__ bias,          // length N, or nullptr
    float* __restrict__ C, int ldc,
    int M, int N, int K, int act, int accum)
{
  __shared__ __align__(16) unsigned short As[BM * AP];   // 64 x 64 bf16
  __shared__ __align__(16) unsigned short Ws[BK * WP];   // 64 x 64 bf16 (K-major)

  const int tid  = threadIdx.x;
  const int lane = tid & 31;
  const int wave = tid >> 5;
  const int m0 = blockIdx.y * BM;
  const int n0 = blockIdx.x * BN;

  const int wm = (wave & 3) * 16;   // wave M offset inside tile
  const int wn = (wave >> 2) * 32;  // wave N offset inside tile

  v8f acc0 = {}; v8f acc1 = {};

  // A staging coords: 64 rows x 64 cols f32; 16 f32 per thread
  const int arow = tid >> 2;            // 0..63
  const int acol = (tid & 3) * 16;      // 0,16,32,48

  for (int k0 = 0; k0 < K; k0 += BK) {
    // ---- async-stage B tile: Ws[k][n] <- Wt[k0+k][n0+n], contiguous 16B rows
    {
      #pragma unroll
      for (int h = 0; h < 2; ++h) {
        const int c   = tid + h * 256;
        const int row = c >> 3;
        const int sub = (c & 7) * 8;
        copy16_to_lds(&Ws[row * WP + sub],
                      Wt + (size_t)(k0 + row) * N + n0 + sub);
      }
    }
    // ---- stage A tile (convert f32 -> bf16) ----
    {
      union { unsigned short s[16]; uint4 u[2]; } tmp;
      const int gr = m0 + arow;
      if (gr < M) {
        const float4* src = (const float4*)(A + (size_t)gr * lda + k0 + acol);
        #pragma unroll
        for (int q = 0; q < 4; ++q) {
          float4 a = src[q];
          tmp.s[q * 4 + 0] = f32_bf16(a.x); tmp.s[q * 4 + 1] = f32_bf16(a.y);
          tmp.s[q * 4 + 2] = f32_bf16(a.z); tmp.s[q * 4 + 3] = f32_bf16(a.w);
        }
        if (k0 + BK < K)   // global_prefetch_b8 for next K slab
          __builtin_prefetch(A + (size_t)gr * lda + k0 + BK + acol, 0, 3);
      } else {
        #pragma unroll
        for (int j = 0; j < 16; ++j) tmp.s[j] = 0;
      }
      *(uint4*)&As[arow * AP + acol]     = tmp.u[0];
      *(uint4*)&As[arow * AP + acol + 8] = tmp.u[1];
    }
    wait_async_copies();
    __syncthreads();

    // ---- 2 K-substeps of 32, 2 WMMAs each => 4 WMMAs per sync period ----
    #pragma unroll
    for (int s = 0; s < BK; s += 32) {
      FragB fa, fb0, fb1;
      // A: lane holds row (lane&15); K halves split across lane halves.
      const int r  = wm + (lane & 15);
      const int kh = (lane & 16) ? 8 : 0;
      fa.u[0] = *(const uint4*)&As[r * AP + s + kh];        // K = s+kh .. +7
      fa.u[1] = *(const uint4*)&As[r * AP + s + kh + 16];   // K = s+kh+16 .. +23
      // B: lane holds column (lane&15); K = s+kh..+7 then s+kh+16..+23, matching A.
      const int bn = wn + (lane & 15);
      #pragma unroll
      for (int j = 0; j < 16; ++j) {
        const int kk = s + kh + (j & 7) + ((j >> 3) << 4);
        ((unsigned short*)&fb0)[j] = Ws[kk * WP + bn];
        ((unsigned short*)&fb1)[j] = Ws[kk * WP + bn + 16];
      }

      acc0 = __builtin_amdgcn_wmma_f32_16x16x32_bf16(false, fa.v, false, fb0.v,
                                                     (short)0, acc0, false, false);
      acc1 = __builtin_amdgcn_wmma_f32_16x16x32_bf16(false, fa.v, false, fb1.v,
                                                     (short)0, acc1, false, false);
    }
    __syncthreads();
  }

  // ---- epilogue: bias + (accumulate) + activation ----
  const int cn  = lane & 15;
  const int cmb = (lane & 16) ? 8 : 0;
  #pragma unroll
  for (int v = 0; v < 8; ++v) {
    const int gm  = m0 + wm + cmb + v;
    const int gn0 = n0 + wn + cn;
    const int gn1 = gn0 + 16;
    if (gm < M) {
      if (gn0 < N) {
        float x = acc0[v] + (bias ? bias[gn0] : 0.f);
        float* p = C + (size_t)gm * ldc + gn0;
        if (accum) x += *p;
        if (act == 1) x = tanhf(x); else if (act == 2) x = sigmoidf_(x);
        *p = x;
      }
      if (gn1 < N) {
        float x = acc1[v] + (bias ? bias[gn1] : 0.f);
        float* p = C + (size_t)gm * ldc + gn1;
        if (accum) x += *p;
        if (act == 1) x = tanhf(x); else if (act == 2) x = sigmoidf_(x);
        *p = x;
      }
    }
  }
}

// ---------------------------------------------------------------------------
// avg[b][d] = mean_p img[b][p][d]
// ---------------------------------------------------------------------------
__global__ void mean_kernel(const float* __restrict__ img, float* __restrict__ avg) {
  int id = blockIdx.x * 256 + threadIdx.x;
  if (id >= B_ * D_) return;
  int b = id / D_, d = id - b * D_;
  const float* p = img + (size_t)b * P_ * D_ + d;
  float s = 0.f;
  for (int pp = 0; pp < P_; ++pp) s += p[(size_t)pp * D_];
  avg[id] = s * (1.f / P_);
}

// ---------------------------------------------------------------------------
// Attention: e -> softmax -> alpha (written to output) -> context
// ---------------------------------------------------------------------------
__global__ __launch_bounds__(256) void attention_kernel(
    const float* __restrict__ img,        // B,P,D
    const float* __restrict__ feat_proj,  // B,P,E
    const float* __restrict__ att_h,      // B,E
    const float* __restrict__ v_att,      // E
    const float* __restrict__ b_vatt,     // scalar
    float* __restrict__ context,          // B,D
    float* __restrict__ alphas,           // B,T,P
    int t)
{
  __shared__ float ah[E_];
  __shared__ float as[256];
  __shared__ float red[256];
  const int b = blockIdx.x, tid = threadIdx.x;

  for (int e = tid; e < E_; e += 256) ah[e] = att_h[b * E_ + e];
  __syncthreads();

  float ev = -1e30f;
  if (tid < P_) {
    const float* fp = feat_proj + ((size_t)b * P_ + tid) * E_;
    float acc = b_vatt[0];
    for (int e = 0; e < E_; ++e) acc += v_att[e] * tanhf(fp[e] + ah[e]);
    ev = acc;
  }
  red[tid] = ev;
  __syncthreads();
  for (int s = 128; s > 0; s >>= 1) {
    if (tid < s) red[tid] = fmaxf(red[tid], red[tid + s]);
    __syncthreads();
  }
  const float mx = red[0];
  __syncthreads();
  float ex = (tid < P_) ? __expf(ev - mx) : 0.f;
  red[tid] = ex;
  __syncthreads();
  for (int s = 128; s > 0; s >>= 1) {
    if (tid < s) red[tid] += red[tid + s];
    __syncthreads();
  }
  const float sum = red[0];
  __syncthreads();

  const float alpha = ex / sum;
  as[tid] = alpha;
  if (tid < P_)
    alphas[(size_t)b * T_ * P_ + (size_t)t * P_ + tid] = alpha;
  __syncthreads();

  for (int d = tid; d < D_; d += 256) {
    const float* ip = img + (size_t)b * P_ * D_ + d;
    float s = 0.f;
    for (int p = 0; p < P_; ++p) s += as[p] * ip[(size_t)p * D_];
    context[b * D_ + d] = s;
  }
}

// ---------------------------------------------------------------------------
// lstm_in = [ emb[cap[b][t]] , gate * context ]
// ---------------------------------------------------------------------------
__global__ void pack_kernel(const float* __restrict__ emb,
                            const int* __restrict__ captions,
                            const float* __restrict__ gate,
                            const float* __restrict__ context,
                            float* __restrict__ lstm_in, int t)
{
  const int W = E_ + D_;
  int id = blockIdx.x * 256 + threadIdx.x;
  if (id >= B_ * W) return;
  int b = id / W, col = id - b * W;
  float v;
  if (col < E_) {
    int w = captions[b * T1_ + t];
    v = emb[(size_t)w * E_ + col];
  } else {
    int d = col - E_;
    v = gate[b * D_ + d] * context[b * D_ + d];
  }
  lstm_in[id] = v;
}

// ---------------------------------------------------------------------------
// LSTM pointwise: gates (B,4E) -> updates c,h in place
// ---------------------------------------------------------------------------
__global__ void lstm_kernel(const float* __restrict__ gates,
                            float* __restrict__ h, float* __restrict__ c)
{
  int id = blockIdx.x * 256 + threadIdx.x;
  if (id >= B_ * E_) return;
  int b = id / E_, e = id - b * E_;
  const float* g = gates + (size_t)b * 4 * E_;
  float gi = sigmoidf_(g[e]);
  float gf = sigmoidf_(g[E_ + e]);
  float gg = tanhf(g[2 * E_ + e]);
  float go = sigmoidf_(g[3 * E_ + e]);
  float c2 = gf * c[id] + gi * gg;
  float h2 = go * tanhf(c2);
  c[id] = c2;
  h[id] = h2;
}

// ---------------------------------------------------------------------------
static inline void launch_gemm(hipStream_t s, const float* A, int lda,
                               const unsigned short* Wt, const float* bias,
                               float* C, int ldc, int M, int N, int K,
                               int act, int accum) {
  dim3 g((N + BN - 1) / BN, (M + BM - 1) / BM), b(256);
  hipLaunchKernelGGL(wmma_gemm_xwt, g, b, 0, s, A, lda, Wt, bias, C, ldc,
                     M, N, K, act, accum);
}
static inline void launch_transpose(hipStream_t s, const float* W,
                                    unsigned short* Wt, int N, int K) {
  dim3 g((K + 31) / 32, (N + 31) / 32), b(256);
  hipLaunchKernelGGL(transpose_bf16_kernel, g, b, 0, s, W, Wt, N, K);
}

extern "C" void kernel_launch(void* const* d_in, const int* in_sizes, int n_in,
                              void* d_out, int out_size, void* d_ws, size_t ws_size,
                              hipStream_t stream) {
  const float* img      = (const float*)d_in[0];
  const int*   captions = (const int*)  d_in[1];
  const float* emb      = (const float*)d_in[2];
  const float* W_init_h = (const float*)d_in[3];
  const float* b_init_h = (const float*)d_in[4];
  const float* W_init_c = (const float*)d_in[5];
  const float* b_init_c = (const float*)d_in[6];
  const float* W_fbeta  = (const float*)d_in[7];
  const float* b_fbeta  = (const float*)d_in[8];
  const float* U_att    = (const float*)d_in[9];
  const float* b_Uatt   = (const float*)d_in[10];
  const float* W_att    = (const float*)d_in[11];
  const float* b_Watt   = (const float*)d_in[12];
  const float* v_att    = (const float*)d_in[13];
  const float* b_vatt   = (const float*)d_in[14];
  const float* W_ih     = (const float*)d_in[15];
  const float* b_ih     = (const float*)d_in[16];
  const float* W_hh     = (const float*)d_in[17];
  const float* b_hh     = (const float*)d_in[18];
  const float* W_out    = (const float*)d_in[19];
  const float* b_out    = (const float*)d_in[20];

  float* preds  = (float*)d_out;                       // B,T,V
  float* alphas = preds + (size_t)B_ * T_ * V_;        // B,T,P

  // ---- workspace carve-out ----
  // f32 region (~28.7 MB) first, then bf16 K-major weights (~31.8 MB).
  float* ws        = (float*)d_ws;
  size_t off = 0;
  float* feat_proj = ws + off;  off += (size_t)B_ * P_ * E_;
  float* avg       = ws + off;  off += (size_t)B_ * D_;
  float* h         = ws + off;  off += (size_t)B_ * E_;
  float* c         = ws + off;  off += (size_t)B_ * E_;
  float* att_h     = ws + off;  off += (size_t)B_ * E_;
  float* context   = ws + off;  off += (size_t)B_ * D_;
  float* gate      = ws + off;  off += (size_t)B_ * D_;
  float* lstm_in   = ws + off;  off += (size_t)B_ * (E_ + D_);
  float* gates     = ws + off;  off += (size_t)B_ * 4 * E_;

  unsigned short* bws = (unsigned short*)(ws + off);
  size_t boff = 0;                                     // all counts mult. of 8
  unsigned short* Wt_init_h = bws + boff; boff += (size_t)D_ * E_ + 64;
  unsigned short* Wt_init_c = bws + boff; boff += (size_t)D_ * E_ + 64;
  unsigned short* Wt_att    = bws + boff; boff += (size_t)D_ * E_ + 64;
  unsigned short* Wt_U      = bws + boff; boff += (size_t)E_ * E_ + 64;
  unsigned short* Wt_fbeta  = bws + boff; boff += (size_t)E_ * D_ + 64;
  unsigned short* Wt_ih     = bws + boff; boff += (size_t)(E_ + D_) * 4 * E_ + 64;
  unsigned short* Wt_hh     = bws + boff; boff += (size_t)E_ * 4 * E_ + 64;
  unsigned short* Wt_out    = bws + boff; boff += (size_t)E_ * V_ + 64;
  (void)ws_size; (void)in_sizes; (void)n_in; (void)out_size; (void)boff;

  // ---- one-time weight prep: f32 (N x K) -> bf16 K-major (K x N) ----
  launch_transpose(stream, W_init_h, Wt_init_h, E_,     D_);
  launch_transpose(stream, W_init_c, Wt_init_c, E_,     D_);
  launch_transpose(stream, W_att,    Wt_att,    E_,     D_);
  launch_transpose(stream, U_att,    Wt_U,      E_,     E_);
  launch_transpose(stream, W_fbeta,  Wt_fbeta,  D_,     E_);
  launch_transpose(stream, W_ih,     Wt_ih,     4 * E_, E_ + D_);
  launch_transpose(stream, W_hh,     Wt_hh,     4 * E_, E_);
  launch_transpose(stream, W_out,    Wt_out,    V_,     E_);

  // ---- init: avg, h0 = tanh(avg@W_init_h^T+b), c0 = tanh(avg@W_init_c^T+b) ----
  hipLaunchKernelGGL(mean_kernel, dim3((B_ * D_ + 255) / 256), dim3(256), 0, stream,
                     img, avg);
  launch_gemm(stream, avg, D_, Wt_init_h, b_init_h, h, E_, B_, E_, D_, 1, 0);
  launch_gemm(stream, avg, D_, Wt_init_c, b_init_c, c, E_, B_, E_, D_, 1, 0);

  // ---- feat_proj = img @ W_att^T + b_Watt : (B*P, E) ----
  launch_gemm(stream, img, D_, Wt_att, b_Watt, feat_proj, E_, B_ * P_, E_, D_, 0, 0);

  // ---- recurrent steps ----
  for (int t = 0; t < T_; ++t) {
    launch_gemm(stream, h, E_, Wt_U, b_Uatt, att_h, E_, B_, E_, E_, 0, 0);
    hipLaunchKernelGGL(attention_kernel, dim3(B_), dim3(256), 0, stream,
                       img, feat_proj, att_h, v_att, b_vatt, context, alphas, t);
    launch_gemm(stream, h, E_, Wt_fbeta, b_fbeta, gate, D_, B_, D_, E_, 2, 0);
    hipLaunchKernelGGL(pack_kernel, dim3((B_ * (E_ + D_) + 255) / 256), dim3(256),
                       0, stream, emb, captions, gate, context, lstm_in, t);
    launch_gemm(stream, lstm_in, E_ + D_, Wt_ih, b_ih, gates, 4 * E_, B_, 4 * E_,
                E_ + D_, 0, 0);
    launch_gemm(stream, h, E_, Wt_hh, b_hh, gates, 4 * E_, B_, 4 * E_, E_, 0, 1);
    hipLaunchKernelGGL(lstm_kernel, dim3((B_ * E_ + 255) / 256), dim3(256), 0,
                       stream, gates, h, c);
    launch_gemm(stream, h, E_, Wt_out, b_out, preds + (size_t)t * V_, T_ * V_,
                B_, V_, E_, 0, 0);
  }
}
